// VectorQuantizer_32126355374648
// MI455X (gfx1250) — compile-verified, matrix operand order checked by reading, not run on hardware
//
#include <hip/hip_runtime.h>
#include <hip/hip_bf16.h>

// ---------------------------------------------------------------------------
// VQ (l2-normalized) for MI455X / gfx1250, wave32 + v_wmma_f32_16x16x32_f16.
// N=4096 z-rows, K=16384 codes, D=64.  Score matrix never materialized
// (268MB > 192MB L2): recompute GEMM in 2 passes via split-f16 WMMA
// (hi+lo, 3 products -> ~fp32 accuracy on the f16 matrix pipe).
// Softmax stats kept in log2 domain (exp2f == raw v_exp_f32), branchless
// online updates, two interleaved WMMA accumulator chains.
// ---------------------------------------------------------------------------

typedef __attribute__((ext_vector_type(16))) _Float16 v16h;
typedef __attribute__((ext_vector_type(8)))  float    v8f;

#define N_ROWS  4096
#define N_CODES 16384
#define EDIM    64
#define ZQ_ELEMS 262144          // 4*64*32*32
#define OUT_SCALARS 262144       // vq, commit, entropy, perplexity at [262144..262147]
#define OUT_IDX 262148           // idx (as float) at [262148..266243]

#define K_LOG2  288.5390081777927f   // 200 / ln(2)  (logits in log2 domain)
#define LN2F    0.6931471805599453f

// workspace offsets (bytes)
#define OFF_ZN      0u                       // float  [N,64]
#define OFF_EB      1048576u                 // float  [K,64]
#define OFF_AHI     5242880u                 // f16    [N,64]
#define OFF_ALO     5767168u                 // f16    [N,64]
#define OFF_BHI     6291456u                 // f16    [K,64]
#define OFF_BLO     8388608u                 // f16    [K,64]
#define OFF_MZ      10485760u                // float2 [N]  (row max(log2), 1/Z)
#define OFF_ROWENT  10518528u                // float  [N]  (sum p*logp per row, natural)
#define OFF_IDX     10534912u                // int    [N]
#define OFF_COLSUM  10551296u                // float  [K]  (sum_n p_{nk})
#define OFF_HIST    10616832u                // int    [K]
#define OFF_SCAL    10682368u                // float  [8]  (scal[0] = SSE)

static __device__ __forceinline__ v8f wmma_f16(v16h a, v16h b, v8f c) {
    return __builtin_amdgcn_wmma_f32_16x16x32_f16(false, a, false, b,
                                                  (short)0, c, false, false);
}

// Load one 16-half A/B fragment per ISA 16-bit layout.
// off = (lane>=16 ? 8 : 0) + k0; halves [off..off+7] and [off+16..off+23].
static __device__ __forceinline__ v16h load_frag(const _Float16* rowPtr, int off) {
    union { v16h h; float4 f[2]; } u;
    const float4* p = reinterpret_cast<const float4*>(rowPtr + off);
    u.f[0] = p[0];
    u.f[1] = p[2];
    return u.h;
}

// ---------------------------------------------------------------------------
__global__ void k_init(float* colsum, int* hist, float* scal) {
    int g = blockIdx.x * blockDim.x + threadIdx.x;
    if (g < N_CODES) { colsum[g] = 0.0f; hist[g] = 0; }
    if (g < 8) scal[g] = 0.0f;
}

// normalize z rows (gather from [B,C,H,W]) + fp32->f16 hi/lo split. 1 wave/row.
__global__ void k_norm_z(const float* __restrict__ z, float* __restrict__ zn,
                         _Float16* __restrict__ ahi, _Float16* __restrict__ alo) {
    int n = blockIdx.x, t = threadIdx.x;
    int b = n >> 10, rem = n & 1023, h = rem >> 5, w = rem & 31;
    const float* base = z + ((size_t)b << 16) + (h << 5) + w;
    float v0 = base[(size_t)t << 10];
    float v1 = base[(size_t)(t + 32) << 10];
    float ss = v0 * v0 + v1 * v1;
    #pragma unroll
    for (int mk = 16; mk >= 1; mk >>= 1) ss += __shfl_xor(ss, mk, 32);
    float inv = 1.0f / fmaxf(sqrtf(ss), 1e-12f);
    float a0 = v0 * inv, a1 = v1 * inv;
    int o = n * EDIM;
    zn[o + t] = a0; zn[o + t + 32] = a1;
    _Float16 h0 = (_Float16)a0, h1 = (_Float16)a1;
    ahi[o + t] = h0; ahi[o + t + 32] = h1;
    alo[o + t]      = (_Float16)(a0 - (float)h0);
    alo[o + t + 32] = (_Float16)(a1 - (float)h1);
}

// normalize codebook rows (contiguous) + split. 1 wave/row.
__global__ void k_norm_e(const float* __restrict__ cb, float* __restrict__ eb,
                         _Float16* __restrict__ bhi, _Float16* __restrict__ blo) {
    int n = blockIdx.x, t = threadIdx.x;
    const float* row = cb + (size_t)n * EDIM;
    float v0 = row[t], v1 = row[t + 32];
    float ss = v0 * v0 + v1 * v1;
    #pragma unroll
    for (int mk = 16; mk >= 1; mk >>= 1) ss += __shfl_xor(ss, mk, 32);
    float inv = 1.0f / fmaxf(sqrtf(ss), 1e-12f);
    float a0 = v0 * inv, a1 = v1 * inv;
    size_t o = (size_t)n * EDIM;
    eb[o + t] = a0; eb[o + t + 32] = a1;
    _Float16 h0 = (_Float16)a0, h1 = (_Float16)a1;
    bhi[o + t] = h0; bhi[o + t + 32] = h1;
    blo[o + t]      = (_Float16)(a0 - (float)h0);
    blo[o + t + 32] = (_Float16)(a1 - (float)h1);
}

// Pass 1: per 16-row strip, sweep all K codes: online softmax stats + argmax.
// Block = 8 waves; wave w handles column tiles w, w+8, ...
// Stats: m = running max (log2 units), Z = sum 2^(x-m), S = sum 2^(x-m)*(x-m).
__global__ void __launch_bounds__(256)
k_pass1(const _Float16* __restrict__ ahi, const _Float16* __restrict__ alo,
        const _Float16* __restrict__ bhi, const _Float16* __restrict__ blo,
        float2* __restrict__ mz, float* __restrict__ rowent,
        int* __restrict__ idxArr, int* __restrict__ hist, float* __restrict__ outF) {
    __shared__ float lm[8][16], lZ[8][16], lS[8][16], lB[8][16];
    __shared__ int   lI[8][16];
    int lane = threadIdx.x & 31, wid = threadIdx.x >> 5;
    int mrow = lane & 15, hiH = lane >> 4, adj = hiH * 8;
    int rowBase = blockIdx.x * 16;
    const _Float16* aHr = ahi + (size_t)(rowBase + mrow) * EDIM;
    const _Float16* aLr = alo + (size_t)(rowBase + mrow) * EDIM;
    v16h aH0 = load_frag(aHr, adj), aH1 = load_frag(aHr, adj + 32);
    v16h aL0 = load_frag(aLr, adj), aL1 = load_frag(aLr, adj + 32);

    float m[8], Z[8], S[8], best[8]; int bidx[8];
    #pragma unroll
    for (int r = 0; r < 8; r++) { m[r] = -1e30f; Z[r] = 0.f; S[r] = 0.f; best[r] = -1e30f; bidx[r] = 0; }

    for (int ct = wid; ct < N_CODES / 16; ct += 8) {
        int col = ct * 16 + mrow;
        const _Float16* bHr = bhi + (size_t)col * EDIM;
        const _Float16* bLr = blo + (size_t)col * EDIM;
        v16h bH0 = load_frag(bHr, adj), bH1 = load_frag(bHr, adj + 32);
        v16h bL0 = load_frag(bLr, adj), bL1 = load_frag(bLr, adj + 32);
        // two independent accumulator chains, interleaved (no same-D RAW chain)
        v8f acc0 = {}, acc1 = {};
        acc0 = wmma_f16(aH0, bH0, acc0); acc1 = wmma_f16(aH1, bH1, acc1);
        acc0 = wmma_f16(aH0, bL0, acc0); acc1 = wmma_f16(aH1, bL1, acc1);
        acc0 = wmma_f16(aL0, bH0, acc0); acc1 = wmma_f16(aL1, bH1, acc1);
        #pragma unroll
        for (int r = 0; r < 8; r++) {
            float x = K_LOG2 * (acc0[r] + acc1[r]);      // logit, log2 domain
            bool gt = x > best[r];                        // strict > => first index
            best[r] = gt ? x : best[r];
            bidx[r] = gt ? col : bidx[r];
            float M  = fmaxf(m[r], x);
            float d1 = m[r] - M, d2 = x - M;              // <= 0
            float e1 = exp2f(d1), e2 = exp2f(d2);
            S[r] = fmaf(d2, e2, (S[r] + d1 * Z[r]) * e1);
            Z[r] = fmaf(Z[r], e1, e2);
            m[r] = M;
        }
    }
    // merge across the 16 lanes that share each row (width-16 butterflies)
    #pragma unroll
    for (int mk = 8; mk >= 1; mk >>= 1) {
        #pragma unroll
        for (int r = 0; r < 8; r++) {
            float m2 = __shfl_xor(m[r], mk, 16);
            float Z2 = __shfl_xor(Z[r], mk, 16);
            float S2 = __shfl_xor(S[r], mk, 16);
            float b2 = __shfl_xor(best[r], mk, 16);
            int   i2 = __shfl_xor(bidx[r], mk, 16);
            float M  = fmaxf(m[r], m2);
            float d1 = m[r] - M, d2 = m2 - M;
            float e1 = exp2f(d1), e2 = exp2f(d2);
            S[r] = (S[r] + d1 * Z[r]) * e1 + (S2 + d2 * Z2) * e2;
            Z[r] = Z[r] * e1 + Z2 * e2;
            m[r] = M;
            bool take = (b2 > best[r]) || (b2 == best[r] && i2 < bidx[r]);
            best[r] = take ? b2 : best[r];
            bidx[r] = take ? i2 : bidx[r];
        }
    }
    if ((lane & 15) == 0) {
        #pragma unroll
        for (int r = 0; r < 8; r++) {
            int row = r + 8 * hiH;
            lm[wid][row] = m[r]; lZ[wid][row] = Z[r]; lS[wid][row] = S[r];
            lB[wid][row] = best[r]; lI[wid][row] = bidx[r];
        }
    }
    __syncthreads();
    if (threadIdx.x < 16) {
        int t = threadIdx.x;
        float M = lm[0][t], Zf = lZ[0][t], Sf = lS[0][t], Bb = lB[0][t]; int Bi = lI[0][t];
        for (int w = 1; w < 8; w++) {
            float m2 = lm[w][t], Z2 = lZ[w][t], S2 = lS[w][t], b2 = lB[w][t]; int i2 = lI[w][t];
            float Mn = fmaxf(M, m2);
            float d1 = M - Mn, d2 = m2 - Mn;
            float e1 = exp2f(d1), e2 = exp2f(d2);
            Sf = (Sf + d1 * Zf) * e1 + (S2 + d2 * Z2) * e2;
            Zf = Zf * e1 + Z2 * e2;
            M = Mn;
            bool take = (b2 > Bb) || (b2 == Bb && i2 < Bi);
            Bb = take ? b2 : Bb;
            Bi = take ? i2 : Bi;
        }
        int n = rowBase + t;
        mz[n] = make_float2(M, 1.0f / Zf);
        rowent[n] = (Sf / Zf) * LN2F - logf(Zf);   // sum p*log p (natural) for row n
        idxArr[n] = Bi;
        outF[OUT_IDX + n] = (float)Bi;
        atomicAdd(&hist[Bi], 1);
    }
}

// Pass 2: each wave owns one 16-column tile exclusively; recompute dots
// (identical WMMA sequence) and accumulate column sums of probs. No atomics.
__global__ void __launch_bounds__(256)
k_pass2(const _Float16* __restrict__ ahi, const _Float16* __restrict__ alo,
        const _Float16* __restrict__ bhi, const _Float16* __restrict__ blo,
        const float2* __restrict__ mz, float* __restrict__ colsum) {
    int lane = threadIdx.x & 31, wid = threadIdx.x >> 5;
    int mrow = lane & 15, hiH = lane >> 4, adj = hiH * 8;
    int ct = blockIdx.x * 8 + wid;
    int col = ct * 16 + mrow;
    const _Float16* bHr = bhi + (size_t)col * EDIM;
    const _Float16* bLr = blo + (size_t)col * EDIM;
    v16h bH0 = load_frag(bHr, adj), bH1 = load_frag(bHr, adj + 32);
    v16h bL0 = load_frag(bLr, adj), bL1 = load_frag(bLr, adj + 32);
    float csum = 0.0f;
    for (int rt = 0; rt < N_ROWS / 16; rt++) {
        int rowBase = rt * 16;
        const _Float16* aHr = ahi + (size_t)(rowBase + mrow) * EDIM;
        const _Float16* aLr = alo + (size_t)(rowBase + mrow) * EDIM;
        v16h aH0 = load_frag(aHr, adj), aH1 = load_frag(aHr, adj + 32);
        v16h aL0 = load_frag(aLr, adj), aL1 = load_frag(aLr, adj + 32);
        v8f acc0 = {}, acc1 = {};
        acc0 = wmma_f16(aH0, bH0, acc0); acc1 = wmma_f16(aH1, bH1, acc1);
        acc0 = wmma_f16(aH0, bL0, acc0); acc1 = wmma_f16(aH1, bL1, acc1);
        acc0 = wmma_f16(aL0, bH0, acc0); acc1 = wmma_f16(aL1, bH1, acc1);
        #pragma unroll
        for (int r = 0; r < 8; r++) {
            int row = rowBase + r + 8 * hiH;
            float2 v = mz[row];
            float x = K_LOG2 * (acc0[r] + acc1[r]);
            csum = fmaf(exp2f(x - v.x), v.y, csum);
        }
    }
    csum += __shfl_xor(csum, 16, 32);           // combine row halves
    if (lane < 16) colsum[ct * 16 + lane] = csum;
}

// gather z_q = emb[idx] into [B,C,H,W] output + SSE for vq/commit loss
__global__ void __launch_bounds__(256)
k_gather(const float* __restrict__ zn, const float* __restrict__ eb,
         const int* __restrict__ idxArr, float* __restrict__ outF,
         float* __restrict__ scal) {
    int tid = threadIdx.x;
    int n = blockIdx.x * 4 + (tid >> 6);
    int c = tid & 63;
    int j = idxArr[n];
    float q  = eb[(size_t)j * EDIM + c];
    float zz = zn[(size_t)n * EDIM + c];
    float d  = q - zz;
    int b = n >> 10, rem = n & 1023, h = rem >> 5, w = rem & 31;
    outF[((size_t)b << 16) + ((size_t)c << 10) + (h << 5) + w] = q;
    float s = d * d;
    #pragma unroll
    for (int mk = 16; mk >= 1; mk >>= 1) s += __shfl_xor(s, mk, 32);
    if ((tid & 31) == 0) atomicAdd(&scal[0], s);
}

static __device__ float block_sum(float v, float* lds) {
    #pragma unroll
    for (int mk = 16; mk >= 1; mk >>= 1) v += __shfl_xor(v, mk, 32);
    int lane = threadIdx.x & 31, wid = threadIdx.x >> 5;
    __syncthreads();
    if (lane == 0) lds[wid] = v;
    __syncthreads();
    float r = (threadIdx.x < 8) ? lds[threadIdx.x] : 0.0f;
    #pragma unroll
    for (int mk = 4; mk >= 1; mk >>= 1) r += __shfl_xor(r, mk, 32);
    return r;   // valid in thread 0
}

__global__ void __launch_bounds__(256)
k_final(const float* __restrict__ rowent, const float* __restrict__ colsum,
        const int* __restrict__ hist, const float* __restrict__ scal,
        float* __restrict__ outF) {
    __shared__ float lds[8];
    int tid = threadIdx.x;
    const float invN = 1.0f / 4096.0f;
    float s1 = 0.f, s2 = 0.f, s3 = 0.f;
    for (int i = tid; i < N_ROWS; i += 256) s1 += rowent[i];
    for (int i = tid; i < N_CODES; i += 256) {
        float q = colsum[i] * invN;
        s2 += q * logf(q + 1e-5f);                 // avg-entropy term
        float e = (float)hist[i] * invN;
        s3 += e * logf(e + 1e-10f);                // perplexity term
    }
    s1 = block_sum(s1, lds);
    s2 = block_sum(s2, lds);
    s3 = block_sum(s3, lds);
    if (tid == 0) {
        float sse = scal[0];
        float vq  = sse * (1.0f / (float)ZQ_ELEMS);
        float sample_entropy = -s1 * invN;
        float avg_entropy    = -s2;
        outF[OUT_SCALARS + 0] = vq;
        outF[OUT_SCALARS + 1] = 0.25f * vq;                            // BETA
        outF[OUT_SCALARS + 2] = 0.1f * (sample_entropy - avg_entropy); // ENT_RATIO
        outF[OUT_SCALARS + 3] = __expf(-s3);                           // perplexity
    }
}

extern "C" void kernel_launch(void* const* d_in, const int* in_sizes, int n_in,
                              void* d_out, int out_size, void* d_ws, size_t ws_size,
                              hipStream_t stream) {
    const float* z  = (const float*)d_in[0];   // [4,64,32,32]
    const float* cb = (const float*)d_in[1];   // [16384,64]
    float* outF = (float*)d_out;
    char* ws = (char*)d_ws;

    float*    zn     = (float*)   (ws + OFF_ZN);
    float*    eb     = (float*)   (ws + OFF_EB);
    _Float16* ahi    = (_Float16*)(ws + OFF_AHI);
    _Float16* alo    = (_Float16*)(ws + OFF_ALO);
    _Float16* bhi    = (_Float16*)(ws + OFF_BHI);
    _Float16* blo    = (_Float16*)(ws + OFF_BLO);
    float2*   mz     = (float2*)  (ws + OFF_MZ);
    float*    rowent = (float*)   (ws + OFF_ROWENT);
    int*      idxArr = (int*)     (ws + OFF_IDX);
    float*    colsum = (float*)   (ws + OFF_COLSUM);
    int*      hist   = (int*)     (ws + OFF_HIST);
    float*    scal   = (float*)   (ws + OFF_SCAL);

    k_init  <<<64,            256, 0, stream>>>(colsum, hist, scal);
    k_norm_z<<<N_ROWS,         32, 0, stream>>>(z,  zn, ahi, alo);
    k_norm_e<<<N_CODES,        32, 0, stream>>>(cb, eb, bhi, blo);
    k_pass1 <<<N_ROWS / 16,   256, 0, stream>>>(ahi, alo, bhi, blo, mz, rowent,
                                                idxArr, hist, outF);
    k_pass2 <<<N_CODES / 128, 256, 0, stream>>>(ahi, alo, bhi, blo, mz, colsum);
    k_gather<<<N_ROWS / 4,    256, 0, stream>>>(zn, eb, idxArr, outF, scal);
    k_final <<<1,             256, 0, stream>>>(rowent, colsum, hist, scal, outF);
}
